// ContextEncoding_18287970747049
// MI455X (gfx1250) — matrix operand order repaired and checked by reading, not compiled, not run on hardware
//
#include <hip/hip_runtime.h>
#include <hip/hip_bf16.h>
#include <stddef.h>
#include <stdint.h>

// ---------------- problem constants ----------------
static constexpr int BB  = 16;     // batch
static constexpr int CC  = 512;    // channels
static constexpr int HW  = 4096;   // H*W
static constexpr int KK  = 32;     // num codes
static constexpr float kEPS = 1e-5f;

// ---------------- WMMA vector types ----------------
typedef __attribute__((ext_vector_type(8)))  __bf16 v8bf;
typedef __attribute__((ext_vector_type(16))) __bf16 v16bf;
typedef __attribute__((ext_vector_type(8)))  float  v8f;

// ---------------- TDM descriptor vector types ----------------
typedef __attribute__((ext_vector_type(4))) unsigned int u32x4;
typedef __attribute__((ext_vector_type(4))) int          i32x4;
typedef __attribute__((ext_vector_type(8))) int          i32x8;

union Frag16 { v16bf v; v8bf h[2]; };

__device__ __forceinline__ unsigned short f2bf(float f) {
    unsigned u = __float_as_uint(f);
    unsigned r = (u + 0x7FFFu + ((u >> 16) & 1u)) >> 16;   // RNE
    return (unsigned short)r;
}

// A fragment (16x32 bf16, M x K): lane holds row (lane&15);
// lanes 0-15 -> K {0..7,16..23}, lanes 16-31 -> K {8..15,24..31}.
// LDS layout: row-major [rows][32].
__device__ __forceinline__ v16bf lds_fragA(const unsigned short* lds, int row0, int lane) {
    int r = lane & 15, half = lane >> 4;
    const unsigned short* base = lds + (row0 + r) * 32;
    Frag16 f;
    f.h[0] = *(const v8bf*)(base + half * 8);
    f.h[1] = *(const v8bf*)(base + 16 + half * 8);
    return f.v;
}

// B fragment (32x16 bf16, K x N): lane holds col (lane&15);
// lanes 0-15 -> K {0..7,16..23}, lanes 16-31 -> K {8..15,24..31}.
// LDS layout: transposed [cols][32].
__device__ __forceinline__ v16bf lds_fragB(const unsigned short* ldsT, int col0, int lane) {
    int c = lane & 15, half = lane >> 4;
    const unsigned short* base = ldsT + (col0 + c) * 32;
    Frag16 f;
    f.h[0] = *(const v8bf*)(base + half * 8);
    f.h[1] = *(const v8bf*)(base + 16 + half * 8);
    return f.v;
}

#define WMMA_BF16(A, Bf, Cacc) \
    __builtin_amdgcn_wmma_f32_16x16x32_bf16(false, (A), false, (Bf), (short)0, (Cacc), false, false)

// -------- TDM: async 2-D bf16 tile load, Global -> LDS (D# per ISA ch.8) -----
// Toolchain here uses the 6-arg builtin:
//   (uint32x4 g0, int32x8 g1, int32x4 g2, int32x4 g3, int32x8 extra, i32 cpol)
#if __has_builtin(__builtin_amdgcn_tensor_load_to_lds) && \
    __has_builtin(__builtin_amdgcn_s_wait_tensorcnt)
#define USE_TDM 1
__device__ __forceinline__ void tdm_load_2d_bf16(unsigned lds_byte_off,
                                                 const unsigned short* gsrc,
                                                 unsigned tile_w_elems,
                                                 unsigned tile_h_rows,
                                                 unsigned row_stride_elems) {
    unsigned long long ga = (unsigned long long)(uintptr_t)gsrc;
    u32x4 g0;
    g0[0] = 1u;                                            // count=1 (valid), user mode
    g0[1] = lds_byte_off;                                  // D#[63:32]  lds_addr
    g0[2] = (unsigned)(ga & 0xFFFFFFFFu);                  // D#[95:64]  global_addr lo
    g0[3] = (unsigned)((ga >> 32) & 0x01FFFFFFu)           // D#[120:96] global_addr hi
          | (2u << 30);                                    // D#[127:126] type=2 (image)
    i32x8 g1;
    g1[0] = (int)(1u << 16);                               // data_size=1 -> 2 bytes; wg_mask=0
    g1[1] = (int)((row_stride_elems & 0xFFFFu) << 16);     // tensor_dim0[15:0]
    g1[2] = (int)((tile_h_rows & 0xFFFFu) << 16)           // tensor_dim1[15:0]
          | (int)((row_stride_elems >> 16) & 0xFFFFu);     // tensor_dim0[31:16]
    g1[3] = (int)((tile_w_elems & 0xFFFFu) << 16);         // tile_dim0 (tensor_dim1 hi = 0)
    g1[4] = (int)(tile_h_rows & 0xFFFFu);                  // tile_dim1 (tile_dim2 = 0)
    g1[5] = (int)row_stride_elems;                         // tensor_dim0_stride[31:0]
    g1[6] = 0;                                             // stride hi / dim1_stride lo
    g1[7] = 0;                                             // dim1_stride hi
    i32x4 z4 = {0, 0, 0, 0};                               // groups 2/3: zeros (2-D tensor)
    i32x8 z8 = {0, 0, 0, 0, 0, 0, 0, 0};                   // extra group: zeros
    __builtin_amdgcn_tensor_load_to_lds(g0, g1, z4, z4, z8, 0);
}
#else
#define USE_TDM 0
#endif

// ---------------- kernel 0: zero accumulators ----------------
__global__ void k_zero(float* p, int n) {
    int i = blockIdx.x * blockDim.x + threadIdx.x;
    int stride = gridDim.x * blockDim.x;
    for (; i < n; i += stride) p[i] = 0.0f;
}

// ---------------- kernel 1: c2[k] = sum_c codewords[k][c]^2 ----------------
__global__ void k_c2(const float* __restrict__ cw, float* __restrict__ c2) {
    int k = threadIdx.x;
    float s = 0.0f;
    for (int c = 0; c < CC; ++c) { float v = cw[k * CC + c]; s += v * v; }
    c2[k] = s;
}

// ---------------- kernel 2: proj GEMM + BN2 + ReLU + x2, feats -> bf16 ------
// per batch: P(512 x 4096) = W(512x512) @ X(512x4096); tile 128x128, BK=32.
// Double-buffered LDS: one barrier per K-step, staging k+1 overlaps WMMA of k.
__global__ __launch_bounds__(256)
void k_proj_gemm(const float* __restrict__ x, const float* __restrict__ conv_w,
                 const float* __restrict__ g2, const float* __restrict__ b2,
                 const float* __restrict__ m2, const float* __restrict__ v2,
                 unsigned short* __restrict__ featsbf, float* __restrict__ x2) {
    __shared__ unsigned short lA[2][128 * 32];   // W tile  [m][k]
    __shared__ unsigned short lB[2][128 * 32];   // X tile  [n][k] (transposed)

    const int tid = threadIdx.x, lane = tid & 31, wave = tid >> 5;
    const int wy = wave >> 2, wx = wave & 3;            // 2 x 4 waves -> 64x32 each
    const int b = blockIdx.z;
    const int m0 = blockIdx.y * 128, n0 = blockIdx.x * 128;
    const float* Xb = x + (size_t)b * CC * HW;

    auto stage = [&](int buf, int k0) {
        // A: 128x32 f32 -> bf16 (row-major)
        for (int s = 0; s < 4; ++s) {
            int slot = tid + s * 256;                 // 1024 slots
            int m = slot >> 3, k4 = (slot & 7) * 4;
            float4 f = *(const float4*)(conv_w + (size_t)(m0 + m) * CC + k0 + k4);
            unsigned short* d = &lA[buf][m * 32 + k4];
            d[0] = f2bf(f.x); d[1] = f2bf(f.y); d[2] = f2bf(f.z); d[3] = f2bf(f.w);
        }
        // B: 32x128 f32 -> bf16 transposed [n][k]
        for (int s = 0; s < 4; ++s) {
            int slot = tid + s * 256;                 // 1024 slots
            int k = slot >> 5, n4 = (slot & 31) * 4;
            float4 f = *(const float4*)(Xb + (size_t)(k0 + k) * HW + n0 + n4);
            lB[buf][(n4 + 0) * 32 + k] = f2bf(f.x);
            lB[buf][(n4 + 1) * 32 + k] = f2bf(f.y);
            lB[buf][(n4 + 2) * 32 + k] = f2bf(f.z);
            lB[buf][(n4 + 3) * 32 + k] = f2bf(f.w);
        }
    };

    v8f acc[4][2];
    for (int mi = 0; mi < 4; ++mi) for (int ni = 0; ni < 2; ++ni) acc[mi][ni] = (v8f)0.0f;

    stage(0, 0);
    __syncthreads();

    for (int kt = 0; kt < CC / 32; ++kt) {
        const int cur = kt & 1;
        if (kt + 1 < CC / 32) stage(cur ^ 1, (kt + 1) * 32);   // overlaps WMMA below

        v16bf aF[4], bF[2];
        for (int mi = 0; mi < 4; ++mi) aF[mi] = lds_fragA(lA[cur], wy * 64 + mi * 16, lane);
        for (int ni = 0; ni < 2; ++ni) bF[ni] = lds_fragB(lB[cur], wx * 32 + ni * 16, lane);
        for (int mi = 0; mi < 4; ++mi)
            for (int ni = 0; ni < 2; ++ni)
                acc[mi][ni] = WMMA_BF16(aF[mi], bF[ni], acc[mi][ni]);

        __syncthreads();
    }

    // epilogue: BN2 + ReLU, store bf16 feats, accumulate x2
    const int half = lane >> 4, nIdx = lane & 15;
    unsigned short* Fb = featsbf + (size_t)b * CC * HW;
    for (int ni = 0; ni < 2; ++ni) {
        const int p = n0 + wx * 32 + ni * 16 + nIdx;
        float cs = 0.0f;
        for (int mi = 0; mi < 4; ++mi) {
            for (int j = 0; j < 8; ++j) {
                int o = m0 + wy * 64 + mi * 16 + half * 8 + j;
                float s = rsqrtf(v2[o] + kEPS) * g2[o];
                float v = (acc[mi][ni][j] - m2[o]) * s + b2[o];
                v = fmaxf(v, 0.0f);
                Fb[(size_t)o * HW + p] = f2bf(v);
                cs += v * v;
            }
        }
        atomicAdd(&x2[b * HW + p], cs);
    }
}

// ---------------- kernel 3: xc GEMM: (32 x 4096) = codewords(32x512) @ feats ----
__global__ __launch_bounds__(256)
void k_xc_gemm(const float* __restrict__ cw, const unsigned short* __restrict__ featsbf,
               float* __restrict__ xc) {
    __shared__ unsigned short lA[32 * 32];    // codewords tile [m][k]
    __shared__ unsigned short lB[256 * 32];   // feats tile [n][k] (transposed)

    const int tid = threadIdx.x, lane = tid & 31, wave = tid >> 5;
    const int b = blockIdx.z;
    const int n0 = blockIdx.x * 256;
    const unsigned short* Fb = featsbf + (size_t)b * CC * HW;

    v8f acc[2][2];
    for (int mi = 0; mi < 2; ++mi) for (int ni = 0; ni < 2; ++ni) acc[mi][ni] = (v8f)0.0f;

    for (int k0 = 0; k0 < CC; k0 += 32) {
        __syncthreads();
        { // stage A (f32 -> bf16), 256 slots
            int m = tid >> 3, k4 = (tid & 7) * 4;
            float4 f = *(const float4*)(cw + (size_t)m * CC + k0 + k4);
            unsigned short* d = &lA[m * 32 + k4];
            d[0] = f2bf(f.x); d[1] = f2bf(f.y); d[2] = f2bf(f.z); d[3] = f2bf(f.w);
        }
        // stage B (bf16 copy w/ transpose), 2048 slots
        for (int s = 0; s < 8; ++s) {
            int slot = tid + s * 256;
            int k = slot >> 6, n4 = (slot & 63) * 4;
            uint2 u = *(const uint2*)(Fb + (size_t)(k0 + k) * HW + n0 + n4);
            lB[(n4 + 0) * 32 + k] = (unsigned short)(u.x & 0xFFFF);
            lB[(n4 + 1) * 32 + k] = (unsigned short)(u.x >> 16);
            lB[(n4 + 2) * 32 + k] = (unsigned short)(u.y & 0xFFFF);
            lB[(n4 + 3) * 32 + k] = (unsigned short)(u.y >> 16);
        }
        __syncthreads();

        v16bf aF[2], bF[2];
        for (int mi = 0; mi < 2; ++mi) aF[mi] = lds_fragA(lA, mi * 16, lane);
        for (int ni = 0; ni < 2; ++ni) bF[ni] = lds_fragB(lB, wave * 32 + ni * 16, lane);
        for (int mi = 0; mi < 2; ++mi)
            for (int ni = 0; ni < 2; ++ni)
                acc[mi][ni] = WMMA_BF16(aF[mi], bF[ni], acc[mi][ni]);
    }

    const int half = lane >> 4, nIdx = lane & 15;
    for (int mi = 0; mi < 2; ++mi)
        for (int ni = 0; ni < 2; ++ni) {
            int p = n0 + wave * 32 + ni * 16 + nIdx;
            for (int j = 0; j < 8; ++j) {
                int kk = mi * 16 + half * 8 + j;
                xc[((size_t)b * KK + kk) * HW + p] = acc[mi][ni][j];
            }
        }
}

// ---------------- kernel 4: softmax assignment + sum_p assign ----------------
__global__ __launch_bounds__(256)
void k_softmax(const float* __restrict__ xc, const float* __restrict__ x2,
               const float* __restrict__ c2, const float* __restrict__ scale,
               unsigned short* __restrict__ assignbf, float* __restrict__ sumA) {
    __shared__ float ls[KK];
    const int gid = blockIdx.x * blockDim.x + threadIdx.x;   // pixel id (0..65535)
    const int b = gid >> 12, p = gid & (HW - 1);
    if (threadIdx.x < KK) ls[threadIdx.x] = 0.0f;
    __syncthreads();

    float d[KK];
    float x2v = x2[gid];
    float mx = -3.4e38f;
    for (int k = 0; k < KK; ++k) {
        float xcv = xc[((size_t)b * KK + k) * HW + p];
        float dd = scale[k] * (x2v - 2.0f * xcv + c2[k]);
        d[k] = dd;
        mx = fmaxf(mx, dd);
    }
    float s = 0.0f;
    for (int k = 0; k < KK; ++k) { float e = __expf(d[k] - mx); d[k] = e; s += e; }
    float inv = 1.0f / s;

    union { unsigned short u[KK]; uint4 q[KK / 8]; } packed;
    for (int k = 0; k < KK; ++k) {
        float a = d[k] * inv;
        atomicAdd(&ls[k], a);
        packed.u[k] = f2bf(a);
    }
    uint4* dst = (uint4*)(assignbf + (size_t)gid * KK);
    for (int i = 0; i < KK / 8; ++i) dst[i] = packed.q[i];

    __syncthreads();
    if (threadIdx.x < KK) atomicAdd(&sumA[b * KK + threadIdx.x], ls[threadIdx.x]);
}

// ---------------- kernel 5: aggregation GEMM (split-K over pixels) -----------
// per batch: encT(512x32) += feats(512x4096) @ assign(4096x32)
// A tile (pure 2-D bf16 copy) is staged via the Tensor Data Mover.
__global__ __launch_bounds__(256)
void k_agg_gemm(const unsigned short* __restrict__ featsbf,
                const unsigned short* __restrict__ assignbf,
                float* __restrict__ encT) {
    __shared__ unsigned short lA[128 * 32];   // feats tile  [c][p]  (direct copy)
    __shared__ unsigned short lB[32 * 32];    // assign tile [kcode][p] (transposed)

    const int tid = threadIdx.x, lane = tid & 31, wave = tid >> 5;
    const int b = blockIdx.z;
    const int m0 = blockIdx.x * 128;
    const int kbase = blockIdx.y * 512;       // split-K: 512 pixels per block
    const unsigned short* Fb = featsbf + (size_t)b * CC * HW;
    const unsigned short* Ab = assignbf + (size_t)b * HW * KK;

    v8f acc[2];
    acc[0] = (v8f)0.0f; acc[1] = (v8f)0.0f;

    for (int k0 = 0; k0 < 512; k0 += 32) {
        __syncthreads();   // all waves done reading LDS from previous step
#if USE_TDM
        if (wave == 0) {   // wave-uniform branch: single TDM issue per block
            tdm_load_2d_bf16((unsigned)(uintptr_t)&lA[0],
                             Fb + (size_t)m0 * HW + kbase + k0,
                             /*tile_w=*/32, /*tile_h=*/128, /*row_stride=*/HW);
            __builtin_amdgcn_s_wait_tensorcnt(0);
        }
#else
        { // fallback: 128 rows x 32 pixels bf16, 16B chunks
            int m = tid >> 1, ch = (tid & 1) * 16;
            *(uint4*)&lA[m * 32 + ch] =
                *(const uint4*)(Fb + (size_t)(m0 + m) * HW + kbase + k0 + ch);
        }
#endif
        { // stage B: assign [p][kcode] -> lB[kcode][p]
            int p = tid >> 3, n4 = (tid & 7) * 4;
            uint2 u = *(const uint2*)(Ab + (size_t)(kbase + k0 + p) * KK + n4);
            lB[(n4 + 0) * 32 + p] = (unsigned short)(u.x & 0xFFFF);
            lB[(n4 + 1) * 32 + p] = (unsigned short)(u.x >> 16);
            lB[(n4 + 2) * 32 + p] = (unsigned short)(u.y & 0xFFFF);
            lB[(n4 + 3) * 32 + p] = (unsigned short)(u.y >> 16);
        }
        __syncthreads();

        v16bf aF = lds_fragA(lA, wave * 16, lane);
        v16bf bF0 = lds_fragB(lB, 0, lane);
        v16bf bF1 = lds_fragB(lB, 16, lane);
        acc[0] = WMMA_BF16(aF, bF0, acc[0]);
        acc[1] = WMMA_BF16(aF, bF1, acc[1]);
    }

    const int half = lane >> 4, nIdx = lane & 15;
    for (int ni = 0; ni < 2; ++ni)
        for (int j = 0; j < 8; ++j) {
            int c = m0 + wave * 16 + half * 8 + j;
            int kcode = ni * 16 + nIdx;
            atomicAdd(&encT[((size_t)b * CC + c) * KK + kcode], acc[ni][j]);
        }
}

// ---------------- kernel 6: encoded finish + BN1 + ReLU + mean ---------------
__global__ void k_enc_finish(const float* __restrict__ encT, const float* __restrict__ sumA,
                             const float* __restrict__ cw,
                             const float* __restrict__ g1, const float* __restrict__ b1,
                             const float* __restrict__ m1, const float* __restrict__ v1,
                             float* __restrict__ enc_feat) {
    int gid = blockIdx.x * blockDim.x + threadIdx.x;     // (b,c), 8192 total
    int b = gid >> 9, c = gid & (CC - 1);
    float s = 0.0f;
    for (int k = 0; k < KK; ++k) {
        float e = encT[(size_t)gid * KK + k] - sumA[b * KK + k] * cw[k * CC + c];
        float sc = rsqrtf(v1[k] + kEPS) * g1[k];
        e = (e - m1[k]) * sc + b1[k];
        s += fmaxf(e, 0.0f);
    }
    enc_feat[gid] = s * (1.0f / (float)KK);
}

// ---------------- kernel 7: fc + sigmoid gate --------------------------------
__global__ void k_gate(const float* __restrict__ enc_feat, const float* __restrict__ fc_w,
                       const float* __restrict__ fc_b, float* __restrict__ gamma) {
    int gid = blockIdx.x * blockDim.x + threadIdx.x;     // (b,c), 8192 total
    int b = gid >> 9, c = gid & (CC - 1);
    const float* ef = enc_feat + b * CC;
    float acc = fc_b[c];
    for (int i = 0; i < CC; ++i) acc += ef[i] * fc_w[(size_t)c * CC + i];
    gamma[gid] = 1.0f / (1.0f + __expf(-acc));
}

// ---------------- kernel 8: output = relu(x * (1 + gamma)) -------------------
__global__ void k_output(const float* __restrict__ x, const float* __restrict__ gamma,
                         float* __restrict__ out) {
    const int total4 = BB * CC * HW / 4;
    int i = blockIdx.x * blockDim.x + threadIdx.x;
    int stride = gridDim.x * blockDim.x;
    for (; i < total4; i += stride) {
        int idx = i * 4;
        int c = (idx >> 12) & (CC - 1);
        int b = idx >> 21;
        float g = 1.0f + gamma[b * CC + c];
        float4 v = *(const float4*)(x + idx);
        float4 r;
        r.x = fmaxf(v.x * g, 0.0f); r.y = fmaxf(v.y * g, 0.0f);
        r.z = fmaxf(v.z * g, 0.0f); r.w = fmaxf(v.w * g, 0.0f);
        *(float4*)(out + idx) = r;
    }
}

// ---------------- launcher ----------------
extern "C" void kernel_launch(void* const* d_in, const int* in_sizes, int n_in,
                              void* d_out, int out_size, void* d_ws, size_t ws_size,
                              hipStream_t stream) {
    const float* x      = (const float*)d_in[0];
    const float* conv_w = (const float*)d_in[1];
    const float* bn2_g  = (const float*)d_in[2];
    const float* bn2_b  = (const float*)d_in[3];
    const float* bn2_m  = (const float*)d_in[4];
    const float* bn2_v  = (const float*)d_in[5];
    const float* cw     = (const float*)d_in[6];
    const float* scale  = (const float*)d_in[7];
    const float* bn1_g  = (const float*)d_in[8];
    const float* bn1_b  = (const float*)d_in[9];
    const float* bn1_m  = (const float*)d_in[10];
    const float* bn1_v  = (const float*)d_in[11];
    const float* fc_w   = (const float*)d_in[12];
    const float* fc_b   = (const float*)d_in[13];

    // workspace layout (bytes)
    char* ws = (char*)d_ws;
    const size_t featsOff  = 0;                                     // bf16 B*C*HW
    const size_t xcOff     = featsOff + (size_t)BB * CC * HW * 2;   //  67,108,864
    const size_t assignOff = xcOff + (size_t)BB * KK * HW * 4;      // + 8,388,608
    const size_t x2Off     = assignOff + (size_t)BB * HW * KK * 2;  // + 4,194,304
    const size_t sumAOff   = x2Off + (size_t)BB * HW * 4;           // +   262,144
    const size_t encTOff   = sumAOff + (size_t)BB * KK * 4;         // +     2,048
    const size_t gammaOff  = encTOff + (size_t)BB * CC * KK * 4;    // + 1,048,576
    const size_t c2Off     = gammaOff + (size_t)BB * CC * 4;        // +    32,768

    unsigned short* featsbf  = (unsigned short*)(ws + featsOff);
    float*          xc       = (float*)(ws + xcOff);
    unsigned short* assignbf = (unsigned short*)(ws + assignOff);
    float*          x2       = (float*)(ws + x2Off);
    float*          sumA     = (float*)(ws + sumAOff);
    float*          encT     = (float*)(ws + encTOff);
    float*          gamma    = (float*)(ws + gammaOff);
    float*          c2       = (float*)(ws + c2Off);

    float* enc_feat = (float*)d_out;              // (B, C) = 8192 floats
    float* output   = enc_feat + BB * CC;         // (B, C, H, W)

    // x2, sumA, encT are contiguous -> single zero pass
    const int nzero = (BB * HW) + (BB * KK) + (BB * CC * KK);
    k_zero<<<512, 256, 0, stream>>>(x2, nzero);
    k_c2<<<1, KK, 0, stream>>>(cw, c2);

    k_proj_gemm<<<dim3(HW / 128, CC / 128, BB), 256, 0, stream>>>(
        x, conv_w, bn2_g, bn2_b, bn2_m, bn2_v, featsbf, x2);

    k_xc_gemm<<<dim3(HW / 256, 1, BB), 256, 0, stream>>>(cw, featsbf, xc);

    k_softmax<<<(BB * HW) / 256, 256, 0, stream>>>(xc, x2, c2, scale, assignbf, sumA);

    k_agg_gemm<<<dim3(CC / 128, 8, BB), 256, 0, stream>>>(featsbf, assignbf, encT);

    k_enc_finish<<<(BB * CC) / 256, 256, 0, stream>>>(encT, sumA, cw,
                                                      bn1_g, bn1_b, bn1_m, bn1_v, enc_feat);

    k_gate<<<(BB * CC) / 256, 256, 0, stream>>>(enc_feat, fc_w, fc_b, gamma);

    k_output<<<2048, 256, 0, stream>>>(x, gamma, output);
}
